// GPT3Workload_20839181320597
// MI455X (gfx1250) — matrix operand order repaired and checked by reading, not compiled, not run on hardware
//
#include <hip/hip_runtime.h>

typedef __bf16 bf16;
typedef __attribute__((ext_vector_type(16))) __bf16 v16bf;
typedef __attribute__((ext_vector_type(8)))  __bf16 v8bf;
typedef __attribute__((ext_vector_type(4)))  __bf16 v4bf;
typedef __attribute__((ext_vector_type(8)))  float   v8f;
typedef __attribute__((ext_vector_type(4)))  float   v4f;

// ---------------- tiling constants ----------------
#define BM 128
#define BN 128
#define BK 32
#define LDS_PAD 8
#define LDP (BK + LDS_PAD)   // 40 halves = 80B row stride -> conflict-free on 64 banks

constexpr int EPI_F32    = 0;  // store f32 C[m*ldc+n]
constexpr int EPI_BF16   = 1;  // store bf16 C[m*ldc+n]
constexpr int EPI_BF16_T = 2;  // store bf16 C[n*ldc+m]  (transposed, for V^T)

// ---------------------------------------------------------------------------
// Canonical NT GEMM:  C[M,N] = A[M,K] * B[N,K]^T   (A,B bf16, accumulate f32)
// Block tile 128x128, k-step 32. 8 waves: 4 (M) x 2 (N), wave tile 32x64.
// Double-buffered LDS + register staging: global loads for tile k+1 issue
// before the WMMAs of tile k, hiding HBM latency behind the matrix pipes.
// grid = (N/128, M/128, batch)
// ---------------------------------------------------------------------------
template<int EPI>
__global__ __launch_bounds__(256)
void gemm_bf16_nt(const bf16* __restrict__ A, int lda, long long strideA,
                  const bf16* __restrict__ B, int ldb, long long strideB,
                  void* __restrict__ Cv, int ldc, long long strideC,
                  int K)
{
    __shared__ bf16 sA[2][BM * LDP];
    __shared__ bf16 sB[2][BN * LDP];

    const int batch = blockIdx.z;
    A += (long long)batch * strideA;
    B += (long long)batch * strideB;

    const int tid   = threadIdx.x;
    const int lane  = tid & 31;
    const int wave  = tid >> 5;
    const int waveM = (wave >> 1) * 32;   // 0,32,64,96
    const int waveN = (wave & 1) * 64;    // 0,64

    const int m0 = blockIdx.y * BM;
    const int n0 = blockIdx.x * BN;

    // per-thread tile-copy coordinates: 512 16B chunks per tile, 2 per thread
    const int row0 = (tid + 0)   >> 2, kc0 = ((tid + 0)   & 3) * 8;
    const int row1 = (tid + 256) >> 2, kc1 = ((tid + 256) & 3) * 8;

    v8f zero = {};
    v8f acc[2][4];
    #pragma unroll
    for (int i = 0; i < 2; ++i)
        #pragma unroll
        for (int j = 0; j < 4; ++j) acc[i][j] = zero;

    const int l16   = lane & 15;
    const int lhalf = lane >> 4;   // 0 or 1

    v8bf ra0, ra1, rb0, rb1;
    // prologue: fetch tile 0 and park it in LDS buffer 0
    ra0 = *(const v8bf*)&A[(long long)(m0 + row0) * lda + kc0];
    ra1 = *(const v8bf*)&A[(long long)(m0 + row1) * lda + kc1];
    rb0 = *(const v8bf*)&B[(long long)(n0 + row0) * ldb + kc0];
    rb1 = *(const v8bf*)&B[(long long)(n0 + row1) * ldb + kc1];
    *(v8bf*)&sA[0][row0 * LDP + kc0] = ra0;
    *(v8bf*)&sA[0][row1 * LDP + kc1] = ra1;
    *(v8bf*)&sB[0][row0 * LDP + kc0] = rb0;
    *(v8bf*)&sB[0][row1 * LDP + kc1] = rb1;

    const int NK = K / BK;
    int cur = 0;
    for (int kt = 0; kt < NK; ++kt) {
        __syncthreads();   // LDS[cur] visible to all; LDS[cur^1] reads retired

        const bool hasNext = (kt + 1) < NK;
        if (hasNext) {     // issue next tile's global loads early
            const int k1 = (kt + 1) * BK;
            ra0 = *(const v8bf*)&A[(long long)(m0 + row0) * lda + k1 + kc0];
            ra1 = *(const v8bf*)&A[(long long)(m0 + row1) * lda + k1 + kc1];
            rb0 = *(const v8bf*)&B[(long long)(n0 + row0) * ldb + k1 + kc0];
            rb1 = *(const v8bf*)&B[(long long)(n0 + row1) * ldb + k1 + kc1];
        }
        if ((kt + 2) < NK) {  // GL2 prefetch two tiles ahead
            const int k2 = (kt + 2) * BK;
            __builtin_prefetch(&A[(long long)(m0 + row0) * lda + k2 + kc0], 0, 1);
            __builtin_prefetch(&B[(long long)(n0 + row0) * ldb + k2 + kc0], 0, 1);
        }

        // A fragment (16x32 bf16): lanes 0-15 hold K 0..7 & 16..23 of row M=lane,
        // lanes 16-31 hold K 8..15 & 24..31 of row M=lane-16.
        v16bf afrag[2];
        #pragma unroll
        for (int mf = 0; mf < 2; ++mf) {
            const bf16* base = &sA[cur][(waveM + mf * 16 + l16) * LDP + lhalf * 8];
            ((v8bf*)&afrag[mf])[0] = *(const v8bf*)(base);
            ((v8bf*)&afrag[mf])[1] = *(const v8bf*)(base + 16);
        }
        // B fragment (32x16 bf16): lane L holds column N=L%16, K 0..7 & 16..23 (L<16)
        // or K 8..15 & 24..31 (L>=16), matching the A fragment.
        v16bf bfrag[4];
        #pragma unroll
        for (int nf = 0; nf < 4; ++nf) {
            const bf16* base = &sB[cur][(waveN + nf * 16 + l16) * LDP + lhalf * 8];
            ((v8bf*)&bfrag[nf])[0] = *(const v8bf*)(base);
            ((v8bf*)&bfrag[nf])[1] = *(const v8bf*)(base + 16);
        }
        #pragma unroll
        for (int mf = 0; mf < 2; ++mf)
            #pragma unroll
            for (int nf = 0; nf < 4; ++nf)
                acc[mf][nf] = __builtin_amdgcn_wmma_f32_16x16x32_bf16(
                    false, afrag[mf], false, bfrag[nf],
                    (short)0, acc[mf][nf], false, false);

        if (hasNext) {     // drain staged registers into the alternate buffer
            const int nxt = cur ^ 1;
            *(v8bf*)&sA[nxt][row0 * LDP + kc0] = ra0;
            *(v8bf*)&sA[nxt][row1 * LDP + kc1] = ra1;
            *(v8bf*)&sB[nxt][row0 * LDP + kc0] = rb0;
            *(v8bf*)&sB[nxt][row1 * LDP + kc1] = rb1;
        }
        cur ^= 1;
    }

    // C/D layout: VGPR r, lanes 0-15 -> (m = r, n = lane), lanes 16-31 -> (m = r+8)
    const int rOff = lhalf * 8;
    #pragma unroll
    for (int mf = 0; mf < 2; ++mf) {
        #pragma unroll
        for (int nf = 0; nf < 4; ++nf) {
            #pragma unroll
            for (int r = 0; r < 8; ++r) {
                int gm = m0 + waveM + mf * 16 + r + rOff;
                int gn = n0 + waveN + nf * 16 + l16;
                float v = acc[mf][nf][r];
                if (EPI == EPI_F32) {
                    ((float*)Cv)[(long long)batch * strideC + (long long)gm * ldc + gn] = v;
                } else if (EPI == EPI_BF16) {
                    ((bf16*)Cv)[(long long)batch * strideC + (long long)gm * ldc + gn] = (bf16)v;
                } else {
                    ((bf16*)Cv)[(long long)batch * strideC + (long long)gn * ldc + gm] = (bf16)v;
                }
            }
        }
    }
}

// ---------------------------------------------------------------------------
// elementwise f32 -> bf16, 4-wide (n must be a multiple of 4; all sizes are)
// ---------------------------------------------------------------------------
__global__ __launch_bounds__(256)
void cvt_f32_bf16(const float* __restrict__ in, bf16* __restrict__ out, int n4)
{
    int i = blockIdx.x * 256 + threadIdx.x;
    if (i < n4) {
        v4f x = ((const v4f*)in)[i];
        v4bf y;
        #pragma unroll
        for (int j = 0; j < 4; ++j) y[j] = (bf16)x[j];
        ((v4bf*)out)[i] = y;
    }
}

// ---------------------------------------------------------------------------
// transpose-convert: in f32 [rows][cols] -> out bf16 [cols][rows]
// 32x32 LDS tile, block = 256 threads (32x8)
// ---------------------------------------------------------------------------
__global__ __launch_bounds__(256)
void transpose_f32_bf16(const float* __restrict__ in, bf16* __restrict__ out,
                        int rows, int cols)
{
    __shared__ float tile[32][33];
    const int bx = blockIdx.x * 32;   // col tile
    const int by = blockIdx.y * 32;   // row tile
    const int tx = threadIdx.x & 31;
    const int ty = threadIdx.x >> 5;  // 0..7
    #pragma unroll
    for (int i = 0; i < 32; i += 8)
        tile[ty + i][tx] = in[(long long)(by + ty + i) * cols + bx + tx];
    __syncthreads();
    #pragma unroll
    for (int i = 0; i < 32; i += 8)
        out[(long long)(bx + ty + i) * rows + by + tx] = (bf16)tile[tx][ty + i];
}

// ---------------------------------------------------------------------------
// row softmax: in f32 [nrows][ncols] -> out bf16, one block (256 thr) per row
// ---------------------------------------------------------------------------
__global__ __launch_bounds__(256)
void softmax_rows(const float* __restrict__ in, bf16* __restrict__ out, int ncols)
{
    const long long row = blockIdx.x;
    const float* rp = in + row * ncols;
    bf16* op = out + row * ncols;
    __shared__ float redM[8];
    __shared__ float redS[8];
    const int tid = threadIdx.x, lane = tid & 31, wave = tid >> 5;

    float mx = -3.4e38f;
    for (int c = tid; c < ncols; c += 256) mx = fmaxf(mx, rp[c]);
    #pragma unroll
    for (int o = 16; o > 0; o >>= 1) mx = fmaxf(mx, __shfl_xor(mx, o, 32));
    if (lane == 0) redM[wave] = mx;
    __syncthreads();
    float m = redM[0];
    #pragma unroll
    for (int i = 1; i < 8; ++i) m = fmaxf(m, redM[i]);

    float s = 0.f;
    for (int c = tid; c < ncols; c += 256) s += __expf(rp[c] - m);
    #pragma unroll
    for (int o = 16; o > 0; o >>= 1) s += __shfl_xor(s, o, 32);
    if (lane == 0) redS[wave] = s;
    __syncthreads();
    float tot = 0.f;
    #pragma unroll
    for (int i = 0; i < 8; ++i) tot += redS[i];
    float inv = 1.f / tot;

    for (int c = tid; c < ncols; c += 256) op[c] = (bf16)(__expf(rp[c] - m) * inv);
}

// ---------------------------------------------------------------------------
// host orchestration
// ---------------------------------------------------------------------------
extern "C" void kernel_launch(void* const* d_in, const int* in_sizes, int n_in,
                              void* d_out, int out_size, void* d_ws, size_t ws_size,
                              hipStream_t stream)
{
    (void)in_sizes; (void)n_in; (void)out_size; (void)ws_size;
    const int M = 2048, H = 16, E = 128, D = 2048, C = 8192;

    const float* I    = (const float*)d_in[0];
    const float* WV   = (const float*)d_in[1];
    const float* WK   = (const float*)d_in[2];
    const float* WQ   = (const float*)d_in[3];
    const float* WZ   = (const float*)d_in[4];
    const float* WFFA = (const float*)d_in[5];
    const float* WFFB = (const float*)d_in[6];
    float* out = (float*)d_out;

    char* ws = (char*)d_ws;
    size_t off = 0;
    auto alloc = [&](size_t bytes) -> void* {
        void* p = ws + off; off += (bytes + 255) & ~(size_t)255; return p;
    };
    bf16* Ib    = (bf16*)alloc((size_t)M * D * 2);      // I bf16 [M][D]
    bf16* WVb   = (bf16*)alloc((size_t)H * E * D * 2);  // [h][e][d]  (NT-ready)
    bf16* WKb   = (bf16*)alloc((size_t)H * E * D * 2);
    bf16* WQb   = (bf16*)alloc((size_t)H * E * D * 2);
    bf16* WZt   = (bf16*)alloc((size_t)D * D * 2);      // [g][hf]
    bf16* WFFAt = (bf16*)alloc((size_t)C * D * 2);      // [c][g]
    bf16* WFFBt = (bf16*)alloc((size_t)D * C * 2);      // [j][c]
    bf16* Qh    = (bf16*)alloc((size_t)H * M * E * 2);  // [h][m][e]
    bf16* Kh    = (bf16*)alloc((size_t)H * M * E * 2);  // [h][m][e]
    bf16* Vt    = (bf16*)alloc((size_t)H * E * M * 2);  // [h][e][m]
    float* QKb  = (float*)alloc((size_t)M * M * 4);     // per-head scores f32
    bf16* Sb    = (bf16*)alloc((size_t)M * M * 2);      // per-head softmax bf16
    bf16* AVb   = (bf16*)alloc((size_t)M * D * 2);      // [m][h*E+e]
    bf16* Zb    = (bf16*)alloc((size_t)M * D * 2);      // [m][g]
    bf16* FFAb  = (bf16*)alloc((size_t)M * C * 2);      // [m][c]

    // ---- 1. precision conversion -----------------------------------------
    {
        int n4 = (M * D) / 4;
        cvt_f32_bf16<<<(n4 + 255) / 256, 256, 0, stream>>>(I, Ib, n4);
        n4 = (H * E * D) / 4;
        cvt_f32_bf16<<<(n4 + 255) / 256, 256, 0, stream>>>(WV, WVb, n4);
        cvt_f32_bf16<<<(n4 + 255) / 256, 256, 0, stream>>>(WK, WKb, n4);
        cvt_f32_bf16<<<(n4 + 255) / 256, 256, 0, stream>>>(WQ, WQb, n4);
    }
    transpose_f32_bf16<<<dim3(D / 32, D / 32), 256, 0, stream>>>(WZ,   WZt,   D, D);
    transpose_f32_bf16<<<dim3(C / 32, D / 32), 256, 0, stream>>>(WFFA, WFFAt, D, C);
    transpose_f32_bf16<<<dim3(D / 32, C / 32), 256, 0, stream>>>(WFFB, WFFBt, C, D);

    // ---- 2. QKV projections (batched over heads, N=E per head) -----------
    // Q[h][m][e] = I[m,:] . WQ[h][e][:]
    gemm_bf16_nt<EPI_BF16><<<dim3(E / BN, M / BM, H), 256, 0, stream>>>(
        Ib, D, 0, WQb, D, (long long)E * D, Qh, E, (long long)M * E, D);
    gemm_bf16_nt<EPI_BF16><<<dim3(E / BN, M / BM, H), 256, 0, stream>>>(
        Ib, D, 0, WKb, D, (long long)E * D, Kh, E, (long long)M * E, D);
    // V stored transposed per head: Vt[h][e][m]
    gemm_bf16_nt<EPI_BF16_T><<<dim3(E / BN, M / BM, H), 256, 0, stream>>>(
        Ib, D, 0, WVb, D, (long long)E * D, Vt, M, (long long)E * M, D);

    // ---- 3. attention per head (reuse score buffers) ---------------------
    for (int h = 0; h < H; ++h) {
        // scores[m][p] = Q[h][m][:] . K[h][p][:]   (K-dim = E = 128)
        gemm_bf16_nt<EPI_F32><<<dim3(M / BN, M / BM, 1), 256, 0, stream>>>(
            Qh + (long long)h * M * E, E, 0,
            Kh + (long long)h * M * E, E, 0,
            QKb, M, 0, E);
        // softmax over p (row-wise)
        softmax_rows<<<M, 256, 0, stream>>>(QKb, Sb, M);
        // AV[m][h*E+e] = S[m][:] . Vt[h][e][:]   (K-dim = M)
        gemm_bf16_nt<EPI_BF16><<<dim3(E / BN, M / BM, 1), 256, 0, stream>>>(
            Sb, M, 0,
            Vt + (long long)h * E * M, M, 0,
            AVb + (long long)h * E, D, 0, M);
    }

    // ---- 4. output projection + FFN --------------------------------------
    gemm_bf16_nt<EPI_BF16><<<dim3(D / BN, M / BM, 1), 256, 0, stream>>>(
        AVb, D, 0, WZt, D, 0, Zb, D, 0, D);
    gemm_bf16_nt<EPI_BF16><<<dim3(C / BN, M / BM, 1), 256, 0, stream>>>(
        Zb, D, 0, WFFAt, D, 0, FFAb, C, 0, D);
    gemm_bf16_nt<EPI_F32><<<dim3(D / BN, M / BM, 1), 256, 0, stream>>>(
        FFAb, C, 0, WFFBt, C, 0, out, D, 0, C);
}
